// S2CAttention_67422396613084
// MI455X (gfx1250) — matrix operand order repaired and checked by reading, not compiled, not run on hardware
//
#include <hip/hip_runtime.h>
#include <hip/hip_bf16.h>

// ---------------------------------------------------------------------------
// S2CAttention block for MI455X (gfx1250).
//  - All GEMMs: v_wmma_f32_16x16x32_bf16 (bf16 operands, fp32 accumulate).
//  - A panels staged into LDS via TDM (tensor_load_to_lds + s_wait_tensorcnt).
//  - Fragments load as two 16-byte vector loads (A row-major; B N-major).
// ---------------------------------------------------------------------------

typedef __bf16          v16bf __attribute__((ext_vector_type(16)));
typedef float           v8f   __attribute__((ext_vector_type(8)));
typedef unsigned short  v16us __attribute__((ext_vector_type(16)));
typedef unsigned short  v8us  __attribute__((ext_vector_type(8)));
typedef unsigned int    v4ui  __attribute__((ext_vector_type(4)));
typedef int             v8i   __attribute__((ext_vector_type(8)));
typedef int             v4i   __attribute__((ext_vector_type(4)));

#define DEV static __device__ __forceinline__

DEV unsigned short f2bf(float f) {
    unsigned int u = __float_as_uint(f);
    u += 0x7FFFu + ((u >> 16) & 1u);            // round-to-nearest-even
    return (unsigned short)(u >> 16);
}

DEV v8f wmma_bf16(v16bf a, v16bf b, v8f c) {
    return __builtin_amdgcn_wmma_f32_16x16x32_bf16(
        /*neg_a=*/false, a, /*neg_b=*/false, b,
        /*c_mod=*/(short)0, c, /*reuse_a=*/false, /*reuse_b=*/false);
}

DEV v16bf cat8(v8us lo, v8us hi) {
    v16us r = __builtin_shufflevector(lo, hi, 0, 1, 2, 3, 4, 5, 6, 7,
                                              8, 9, 10, 11, 12, 13, 14, 15);
    return __builtin_bit_cast(v16bf, r);
}

// A fragment: 16x32 (MxK) from row-major [rows][lda] bf16-as-u16.
// lane-lo row M holds K {k0..+7, k0+16..+23}; lane-hi K {k0+8..15, k0+24..31}
// => two contiguous 16B chunks per lane.
DEV v16bf load_a(const unsigned short* A, int lda, int m0, int k0, int lane) {
    int row = m0 + (lane & 15);
    int hi8 = (lane & 16) ? 8 : 0;
    const unsigned short* p = A + row * lda + k0 + hi8;
    return cat8(*(const v8us*)p, *(const v8us*)(p + 16));
}

// B fragment: 32x16 (KxN) from an N-major source S[n][ldn] (K contiguous).
DEV v16bf load_bt(const unsigned short* S, int ldn, int k0, int n0, int lane) {
    int col  = n0 + (lane & 15);
    int hi16 = (lane & 16) ? 8 : 0;
    const unsigned short* p = S + col * ldn + k0 + hi16;
    return cat8(*(const v8us*)p, *(const v8us*)(p + 16));
}

// ---------------------------------------------------------------------------
// TDM: async 2D tile load global->LDS (Tensor Data Mover, ISA ch.8).
// D# group0: count=1 | lds_addr[63:32] | global_addr[120:64] | type=2[127:126]
// D# group1: data_size[17:16]=1 (2B) | tensor_dim0[79:48] | tensor_dim1[111:80]
//            | tile_dim0[127:112] | tile_dim1[143:128] | dim0_stride[207:160]
// Completion tracked with TENSORcnt (s_wait_tensorcnt 0).
// This toolchain's builtin is the 6-arg form:
//   (uint32x4 g0, int32x8 g1, int32x4, int32x4, int32x8, i32 cpol)
// ---------------------------------------------------------------------------
DEV void tdm_load_2d(unsigned int lds_addr, const void* gptr,
                     unsigned int td0, unsigned int td1,
                     unsigned int tile0, unsigned int tile1,
                     unsigned int stride0) {
    unsigned long long ga = (unsigned long long)(size_t)gptr;
    v4ui g0;
    g0[0] = 1u;                                               // count=1, user D#
    g0[1] = lds_addr;                                         // LDS byte address
    g0[2] = (unsigned int)ga;                                 // global_addr[31:0]
    g0[3] = (unsigned int)((ga >> 32) & 0x01FFFFFFu) | (2u << 30); // addr hi | type=2
    v8i g1;
    g1[0] = (int)(1u << 16);                                  // data_size=2 bytes
    g1[1] = (int)((td0 & 0xFFFFu) << 16);                     // tensor_dim0 lo
    g1[2] = (int)((td0 >> 16) | ((td1 & 0xFFFFu) << 16));     // dim0 hi | dim1 lo
    g1[3] = (int)((td1 >> 16) | (tile0 << 16));               // dim1 hi | tile_dim0
    g1[4] = (int)(tile1 & 0xFFFFu);                           // tile_dim1, tile_dim2=0
    g1[5] = (int)stride0;                                     // dim0_stride lo
    g1[6] = 0;
    g1[7] = 0;
    v4i gz4 = {0, 0, 0, 0};
    v8i gz8 = {0, 0, 0, 0, 0, 0, 0, 0};
    __builtin_amdgcn_tensor_load_to_lds(g0, g1, gz4, gz4, gz8, 0);
}

#define LDS_ADDR(p) ((unsigned int)(size_t)(p))

// ---------------------------------------------------------------------------
// Kernel 0: weight prep -> bf16, N-major (K-contiguous) B layouts.
// ---------------------------------------------------------------------------
__global__ void prep_weights_k(const float* qkv_w, const float* proj_w,
                               const float* fc1_w, const float* fc2_w, const float* fc3_w,
                               unsigned short* qkv_wbT, unsigned short* proj_wbT,
                               unsigned short* w1b, unsigned short* w2b,
                               unsigned short* w3b) {
    int i = blockIdx.x * blockDim.x + threadIdx.x;
    const int N1 = 576 * 192, N2 = 192 * 192, N3 = 768 * 192;
    if (i < N1) { int o = i / 192, c = i % 192; qkv_wbT[i] = f2bf(qkv_w[c * 576 + o]); return; }
    i -= N1;
    if (i < N2) { int o = i / 192, c = i % 192; proj_wbT[i] = f2bf(proj_w[c * 192 + o]); return; }
    i -= N2;
    if (i < N3) { w1b[i] = f2bf(fc1_w[i]); return; }
    i -= N3;
    if (i < N3) { w2b[i] = f2bf(fc2_w[i]); return; }
    i -= N3;
    if (i < N3) { w3b[i] = f2bf(fc3_w[i]); return; }
}

// ---------------------------------------------------------------------------
// Kernel 1: LN1 over channels + "faithful" window partition -> P (fp32), Pb (bf16)
// ---------------------------------------------------------------------------
__global__ __launch_bounds__(256) void ln1_window_k(const float* __restrict__ x,
                                                    const float* __restrict__ g,
                                                    const float* __restrict__ bt,
                                                    float* __restrict__ P,
                                                    unsigned short* __restrict__ Pb) {
    const int h = blockIdx.x & 255;
    const int b = blockIdx.x >> 8;
    const int w = threadIdx.x;
    const float* xp = x + ((size_t)b * 192 * 256 + h) * 256 + w;
    float s = 0.f, s2 = 0.f;
#pragma unroll 4
    for (int c = 0; c < 192; ++c) {
        float v = xp[(size_t)c * 65536];
        s += v; s2 += v * v;
    }
    float mean = s * (1.f / 192.f);
    float var  = s2 * (1.f / 192.f) - mean * mean;
    float rstd = rsqrtf(var + 1e-5f);
    const int h1 = h >> 3, h2 = h & 7;
    for (int c = 0; c < 192; ++c) {
        float v = xp[(size_t)c * 65536];           // x is L2-resident on 2nd pass
        float y = (v - mean) * rstd * g[c] + bt[c];
        size_t j = (((((size_t)b * 24 + (c >> 3)) * 32 + h1) * 8 + (c & 7)) * 8 + h2) * 256 + w;
        P[j]  = y;
        Pb[j] = f2bf(y);
    }
}

// ---------------------------------------------------------------------------
// Kernel 2: QKV GEMM. M=131072, K=192, N=576. Block: 64(M) x 192(N), 8 waves.
// A panel (64x192) via one TDM load; K-loop reads A from LDS, B from global.
// ---------------------------------------------------------------------------
__global__ __launch_bounds__(256) void qkv_gemm_k(const unsigned short* __restrict__ Pb,
                                                  const unsigned short* __restrict__ WbT,
                                                  const float* __restrict__ bias,
                                                  unsigned short* __restrict__ qkvb) {
    __shared__ alignas(16) unsigned short As[64][192];
    const int tid = threadIdx.x, lane = tid & 31, wave = tid >> 5;
    const int T0 = blockIdx.x * 64;
    const int nb = blockIdx.y * 192;
    const int m0 = (wave >> 1) * 16;
    const int nh = (wave & 1) * 96;
    if (wave == 0) {
        tdm_load_2d(LDS_ADDR(&As[0][0]), Pb + (size_t)T0 * 192,
                    /*td0=*/192, /*td1=*/131072, /*tile0=*/192, /*tile1=*/64,
                    /*stride0=*/192);
        __builtin_amdgcn_s_wait_tensorcnt(0);
    }
    __syncthreads();
    v8f acc[6] = {};
#pragma unroll
    for (int k0 = 0; k0 < 192; k0 += 32) {
        v16bf af = load_a(&As[0][0], 192, m0, k0, lane);
#pragma unroll
        for (int t = 0; t < 6; ++t) {
            v16bf bf = load_bt(WbT, 192, k0, nb + nh + t * 16, lane);
            acc[t] = wmma_bf16(af, bf, acc[t]);
        }
    }
    const int hi = (lane >> 4) & 1;
#pragma unroll
    for (int t = 0; t < 6; ++t) {
        int col = nb + nh + t * 16 + (lane & 15);
        float bv = bias[col];
        float sc = (col < 192) ? 0.17677669529663687f : 1.0f;  // 1/sqrt(32) on q
#pragma unroll
        for (int v = 0; v < 8; ++v) {
            int row = T0 + m0 + v + 8 * hi;
            qkvb[(size_t)row * 576 + col] = f2bf((acc[t][v] + bv) * sc);
        }
    }
}

// ---------------------------------------------------------------------------
// Kernel 3: window attention. One block per (window, head). 4 waves.
// q,k tiles via TDM (row-major is what QK^T wants); v staged transposed by hand
// so the P@V B-operand loads as two ds_load_b128 per lane.
// ---------------------------------------------------------------------------
__global__ __launch_bounds__(128) void attn_k(const unsigned short* __restrict__ qkvb,
                                              const float* __restrict__ bias_table,
                                              unsigned short* __restrict__ ob) {
    __shared__ alignas(16) unsigned short qs[64][32];
    __shared__ alignas(16) unsigned short ks[64][32];
    __shared__ alignas(16) unsigned short vsT[32][64];
    __shared__ alignas(16) unsigned short ps[64][64];
    const int tid = threadIdx.x, lane = tid & 31, wave = tid >> 5;
    const int win = blockIdx.x, head = blockIdx.y;
    const size_t base = (size_t)win * 64 * 576 + head * 32;
    if (wave == 0) {
        tdm_load_2d(LDS_ADDR(&qs[0][0]), qkvb + base,
                    576, 131072, 32, 64, 576);
        tdm_load_2d(LDS_ADDR(&ks[0][0]), qkvb + base + 192,
                    576, 131072, 32, 64, 576);
        __builtin_amdgcn_s_wait_tensorcnt(0);
    }
    {
        int r = tid >> 1, hf = (tid & 1) * 16;     // 128 thr = 64 rows x 2 halves
        size_t ro = base + (size_t)r * 576 + hf + 384;
        v8us va = *(const v8us*)&qkvb[ro];
        v8us vb = *(const v8us*)&qkvb[ro + 8];
#pragma unroll
        for (int j = 0; j < 8; ++j) {
            vsT[hf + j][r]     = va[j];
            vsT[hf + 8 + j][r] = vb[j];
        }
    }
    __syncthreads();
    const int m0 = wave * 16;
    const int hi = (lane >> 4) & 1;
    const int cb = lane & 15;
    v8f zro = {};
    v8f s[4];
    v16bf af = load_a(&qs[0][0], 32, m0, 0, lane);
#pragma unroll
    for (int nt = 0; nt < 4; ++nt) {
        v16bf bf = load_bt(&ks[0][0], 32, 0, nt * 16, lane);   // B[d][m] = k[m][d]
        s[nt] = wmma_bf16(af, bf, zro);
    }
    // relative position bias: bias_table[((i1-i2+7)*15 + (j1-j2+7)) * 6 + head]
#pragma unroll
    for (int nt = 0; nt < 4; ++nt) {
        int m = nt * 16 + cb;
        int i2 = m >> 3, j2 = m & 7;
#pragma unroll
        for (int v = 0; v < 8; ++v) {
            int n = m0 + v + 8 * hi;
            int idx = ((n >> 3) - i2 + 7) * 15 + ((n & 7) - j2 + 7);
            s[nt][v] += bias_table[idx * 6 + head];
        }
    }
    // row softmax: each row lives on one 16-lane half, 4 values per lane
#pragma unroll
    for (int v = 0; v < 8; ++v) {
        float mx = fmaxf(fmaxf(s[0][v], s[1][v]), fmaxf(s[2][v], s[3][v]));
#pragma unroll
        for (int d = 1; d < 16; d <<= 1) mx = fmaxf(mx, __shfl_xor(mx, d, 32));
        float e0 = __expf(s[0][v] - mx), e1 = __expf(s[1][v] - mx);
        float e2 = __expf(s[2][v] - mx), e3 = __expf(s[3][v] - mx);
        float sm = e0 + e1 + e2 + e3;
#pragma unroll
        for (int d = 1; d < 16; d <<= 1) sm += __shfl_xor(sm, d, 32);
        float inv = 1.0f / sm;
        s[0][v] = e0 * inv; s[1][v] = e1 * inv; s[2][v] = e2 * inv; s[3][v] = e3 * inv;
    }
#pragma unroll
    for (int nt = 0; nt < 4; ++nt) {
        int m = nt * 16 + cb;
#pragma unroll
        for (int v = 0; v < 8; ++v)
            ps[m0 + v + 8 * hi][m] = f2bf(s[nt][v]);
    }
    __syncthreads();
    // O = P @ V : M=16 rows/wave, N=32, K=64 (2 wmma k-steps)
#pragma unroll
    for (int nt = 0; nt < 2; ++nt) {
        v8f o = {};
#pragma unroll
        for (int kk = 0; kk < 2; ++kk) {
            v16bf pa = load_a(&ps[0][0], 64, m0, kk * 32, lane);
            v16bf vb = load_bt(&vsT[0][0], 64, kk * 32, nt * 16, lane);
            o = wmma_bf16(pa, vb, o);
        }
#pragma unroll
        for (int v = 0; v < 8; ++v) {
            int row = m0 + v + 8 * hi;
            ob[((size_t)win * 64 + row) * 192 + head * 32 + nt * 16 + cb] = f2bf(o[v]);
        }
    }
}

// ---------------------------------------------------------------------------
// Kernel 4: proj GEMM + residual (P) + LN2, fused. Block: 64 tokens x full 192 N.
// A panel via TDM; A (24KB) and LN buffer yb (48KB) share one LDS region
// (disjoint lifetimes, barrier-separated).
// ---------------------------------------------------------------------------
__global__ __launch_bounds__(256) void proj_ln2_k(const unsigned short* __restrict__ ob,
                                                  const unsigned short* __restrict__ WbT,
                                                  const float* __restrict__ pbias,
                                                  const float* __restrict__ P,
                                                  const float* __restrict__ g2,
                                                  const float* __restrict__ b2,
                                                  float* __restrict__ zf,
                                                  unsigned short* __restrict__ zb) {
    __shared__ alignas(16) unsigned char smem[64 * 192 * 4];   // 48KB overlay
    unsigned short (*As)[192] = (unsigned short(*)[192])smem;  // 24KB (phase 1)
    float          (*yb)[192] = (float(*)[192])smem;           // 48KB (phase 2)
    const int tid = threadIdx.x, lane = tid & 31, wave = tid >> 5;
    const int T0 = blockIdx.x * 64;
    const int m0 = (wave >> 1) * 16;
    const int nh = (wave & 1) * 96;
    if (wave == 0) {
        tdm_load_2d(LDS_ADDR(&As[0][0]), ob + (size_t)T0 * 192,
                    192, 131072, 192, 64, 192);
        __builtin_amdgcn_s_wait_tensorcnt(0);
    }
    __syncthreads();
    v8f acc[6] = {};
#pragma unroll
    for (int k0 = 0; k0 < 192; k0 += 32) {
        v16bf af = load_a(&As[0][0], 192, m0, k0, lane);
#pragma unroll
        for (int t = 0; t < 6; ++t) {
            v16bf bf = load_bt(WbT, 192, k0, nh + t * 16, lane);
            acc[t] = wmma_bf16(af, bf, acc[t]);
        }
    }
    __syncthreads();                                // A panel dead; reuse as yb
    const int hi = (lane >> 4) & 1;
#pragma unroll
    for (int t = 0; t < 6; ++t) {
        int col = nh + t * 16 + (lane & 15);
        float bv = pbias[col];
#pragma unroll
        for (int v = 0; v < 8; ++v) {
            int row = m0 + v + 8 * hi;
            yb[row][col] = acc[t][v] + bv + P[(size_t)(T0 + row) * 192 + col];
        }
    }
    __syncthreads();
    // LN2: thread -> (row = tid/4, 48-col strip = tid%4); reduce across 4 lanes
    {
        int r = tid >> 2, q = tid & 3, c0 = q * 48;
        float s = 0.f, s2 = 0.f;
#pragma unroll
        for (int c = 0; c < 48; ++c) {
            float v = yb[r][c0 + c]; s += v; s2 += v * v;
        }
#pragma unroll
        for (int d = 1; d < 4; d <<= 1) {
            s  += __shfl_xor(s,  d, 32);
            s2 += __shfl_xor(s2, d, 32);
        }
        float mean = s * (1.f / 192.f);
        float var  = s2 * (1.f / 192.f) - mean * mean;
        float rstd = rsqrtf(var + 1e-5f);
        for (int c = 0; c < 48; ++c) {
            int cc = c0 + c;
            float zv = (yb[r][cc] - mean) * rstd * g2[cc] + b2[cc];
            size_t o = (size_t)(T0 + r) * 192 + cc;
            zf[o] = zv;
            zb[o] = f2bf(zv);
        }
    }
}

// ---------------------------------------------------------------------------
// Kernel 5: gated FFN fused: h1=sigmoid(z@fc1^T+b1), h2=z@fc2^T+b2, g=h1*h2 (LDS),
// out = g@fc3^T + b3 + z, scattered back to BCHW. Block: 32 tokens, 8 waves.
// z panel via TDM.
// ---------------------------------------------------------------------------
__global__ __launch_bounds__(256) void ffn_k(const unsigned short* __restrict__ zb,
                                             const float* __restrict__ zf,
                                             const unsigned short* __restrict__ w1b,
                                             const float* __restrict__ b1,
                                             const unsigned short* __restrict__ w2b,
                                             const float* __restrict__ b2,
                                             const unsigned short* __restrict__ w3b,
                                             const float* __restrict__ b3,
                                             float* __restrict__ out) {
    __shared__ alignas(16) unsigned short At[32][192];
    __shared__ alignas(16) unsigned short gl[32][768];
    const int tid = threadIdx.x, lane = tid & 31, wave = tid >> 5;
    const int T0 = blockIdx.x * 32;
    if (wave == 0) {
        tdm_load_2d(LDS_ADDR(&At[0][0]), zb + (size_t)T0 * 192,
                    192, 131072, 192, 32, 192);
        __builtin_amdgcn_s_wait_tensorcnt(0);
    }
    __syncthreads();
    const int hi = (lane >> 4) & 1, cb = lane & 15;
    // phase 1: fc1 & fc2, 48 N-tiles = 8 waves x 6 iterations
    for (int ni = 0; ni < 6; ++ni) {
        int n0 = (ni * 8 + wave) * 16;
        v8f a10 = {}, a11 = {}, a20 = {}, a21 = {};
#pragma unroll
        for (int k0 = 0; k0 < 192; k0 += 32) {
            v16bf af0 = load_a(&At[0][0], 192, 0,  k0, lane);
            v16bf af1 = load_a(&At[0][0], 192, 16, k0, lane);
            v16bf bf1 = load_bt(w1b, 192, k0, n0, lane);
            v16bf bf2 = load_bt(w2b, 192, k0, n0, lane);
            a10 = wmma_bf16(af0, bf1, a10);
            a11 = wmma_bf16(af1, bf1, a11);
            a20 = wmma_bf16(af0, bf2, a20);
            a21 = wmma_bf16(af1, bf2, a21);
        }
        int col = n0 + cb;
        float bb1 = b1[col], bb2 = b2[col];
#pragma unroll
        for (int v = 0; v < 8; ++v) {
            float h1a = 1.0f / (1.0f + __expf(-(a10[v] + bb1)));
            gl[v + 8 * hi][col] = f2bf(h1a * (a20[v] + bb2));
            float h1b = 1.0f / (1.0f + __expf(-(a11[v] + bb1)));
            gl[16 + v + 8 * hi][col] = f2bf(h1b * (a21[v] + bb2));
        }
    }
    __syncthreads();
    // phase 2: out = g @ fc3^T (K=768), 12 N-tiles over 8 waves (wave-uniform split)
    for (int rep = 0; rep < 2; ++rep) {
        int nt = wave + rep * 8;
        if (nt >= 12) break;                       // uniform per wave, EXEC stays full
        int n0 = nt * 16;
        v8f o0 = {}, o1 = {};
        for (int k0 = 0; k0 < 768; k0 += 32) {
            v16bf af0 = load_a(&gl[0][0], 768, 0,  k0, lane);
            v16bf af1 = load_a(&gl[0][0], 768, 16, k0, lane);
            v16bf bf  = load_bt(w3b, 768, k0, n0, lane);
            o0 = wmma_bf16(af0, bf, o0);
            o1 = wmma_bf16(af1, bf, o1);
        }
        int c = n0 + cb;
        float bb = b3[c];
#pragma unroll
        for (int v = 0; v < 8; ++v) {
#pragma unroll
            for (int mt = 0; mt < 2; ++mt) {
                int row = mt * 16 + v + 8 * hi;
                int T = T0 + row;
                float val = (mt ? o1[v] : o0[v]) + bb + zf[(size_t)T * 192 + c];
                int b  = T >> 16;
                int a1 = (T >> 11) & 31, a2 = (T >> 6) & 31;
                int a3 = (T >> 3) & 7,  a4 = T & 7;
                int hh = a1 * 8 + a3, ww = a2 * 8 + a4;
                out[(((size_t)b * 192 + c) * 256 + hh) * 256 + ww] = val;
            }
        }
    }
}

// ---------------------------------------------------------------------------
extern "C" void kernel_launch(void* const* d_in, const int* in_sizes, int n_in,
                              void* d_out, int out_size, void* d_ws, size_t ws_size,
                              hipStream_t stream) {
    (void)in_sizes; (void)n_in; (void)out_size; (void)ws_size;
    const float* x      = (const float*)d_in[0];
    const float* ln1g   = (const float*)d_in[1];
    const float* ln1b   = (const float*)d_in[2];
    const float* qkv_w  = (const float*)d_in[3];
    const float* qkv_b  = (const float*)d_in[4];
    const float* btab   = (const float*)d_in[5];
    const float* proj_w = (const float*)d_in[6];
    const float* proj_b = (const float*)d_in[7];
    const float* ln2g   = (const float*)d_in[8];
    const float* ln2b   = (const float*)d_in[9];
    const float* fc1_w  = (const float*)d_in[10];
    const float* fc1_b  = (const float*)d_in[11];
    const float* fc2_w  = (const float*)d_in[12];
    const float* fc2_b  = (const float*)d_in[13];
    const float* fc3_w  = (const float*)d_in[14];
    const float* fc3_b  = (const float*)d_in[15];
    float* out = (float*)d_out;

    char* p = (char*)d_ws;
    auto alloc = [&](size_t bytes) -> char* {
        char* r = p;
        p += (bytes + 255) & ~(size_t)255;
        return r;
    };
    const size_t TOT = 25165824ULL;  // B*C*H*W = 131072 tokens x 192
    float*          P        = (float*)alloc(TOT * 4);
    unsigned short* Pb       = (unsigned short*)alloc(TOT * 2);
    unsigned short* qkvb     = (unsigned short*)alloc(131072ULL * 576 * 2);
    unsigned short* ob       = (unsigned short*)alloc(131072ULL * 192 * 2);
    float*          zf       = (float*)alloc(TOT * 4);
    unsigned short* zbuf     = (unsigned short*)alloc(TOT * 2);
    unsigned short* qkv_wbT  = (unsigned short*)alloc(576 * 192 * 2);
    unsigned short* proj_wbT = (unsigned short*)alloc(192 * 192 * 2);
    unsigned short* w1b      = (unsigned short*)alloc(768 * 192 * 2);
    unsigned short* w2b      = (unsigned short*)alloc(768 * 192 * 2);
    unsigned short* w3b      = (unsigned short*)alloc(192 * 768 * 2);

    prep_weights_k<<<dim3(589824 / 256), 256, 0, stream>>>(
        qkv_w, proj_w, fc1_w, fc2_w, fc3_w,
        qkv_wbT, proj_wbT, w1b, w2b, w3b);
    ln1_window_k<<<dim3(512), 256, 0, stream>>>(x, ln1g, ln1b, P, Pb);
    qkv_gemm_k<<<dim3(2048, 3), 256, 0, stream>>>(Pb, qkv_wbT, qkv_b, qkvb);
    attn_k<<<dim3(2048, 6), 128, 0, stream>>>(qkvb, btab, ob);
    proj_ln2_k<<<dim3(2048), 256, 0, stream>>>(ob, proj_wbT, proj_b, P, ln2g, ln2b, zf, zbuf);
    ffn_k<<<dim3(4096), 256, 0, stream>>>(zbuf, zf, w1b, fc1_b, w2b, fc2_b,
                                          w3b, fc3_b, out);
}
